// ResourceAllocation_GNN_429496729990
// MI455X (gfx1250) — matrix operand order repaired and checked by reading, not compiled, not run on hardware
//
#include <hip/hip_runtime.h>
#include <hip/hip_bf16.h>

// ---------------- problem constants ----------------
#define B_N      16384
#define IAB_N    10
#define MAXUE_N  10
#define TF_N     6
#define EMB_N    30
#define NN       (B_N * IAB_N)     // 163840 graph nodes
#define NE       (B_N * 40)        // 655360 edges
#define RR       (B_N * IAB_N)     // 163840 head rows
#define XSTRIDE  (60 + TF_N * MAXUE_N * IAB_N)  // 660

// head GEMM tiling
#define KP    320      // padded K (301 -> 320)
#define KT    10       // k tiles of 32
#define NT1   20       // n tiles (301 -> 320) for GEMM1/2
#define NT3   2        // n tiles (22 -> 32) for GEMM3
#define ROWS_WG 128
#define HEAD_THREADS 256

// LDS layout for k_head
#define HP       328                         // padded LDS row stride (bank spread)
#define HBYTES   (ROWS_WG * HP * 2)          // 83968
#define BKHALFS  (NT1 * 32 * 16)             // 10240 halfs = one kt slice of W1p/W2p
#define BKBYTES  (BKHALFS * 2)               // 20480
#define PBYTES   (ROWS_WG * 32 * 4)          // 16384
#define SMEM_BYTES (HBYTES + 2 * BKBYTES + PBYTES)   // 141312

typedef __attribute__((ext_vector_type(8)))  _Float16 v8h;
typedef __attribute__((ext_vector_type(16))) _Float16 v16h;
typedef __attribute__((ext_vector_type(8)))  float    v8f;
typedef __attribute__((ext_vector_type(4)))  int      v4i;

#if defined(__gfx1250__) && __has_builtin(__builtin_amdgcn_global_load_async_to_lds_b128)
#define HAVE_ASYNC 1
#else
#define HAVE_ASYNC 0
#endif

__device__ __forceinline__ float gelu_f(float v) {
    return 0.5f * v * (1.0f + erff(v * 0.70710678118654752f));
}

__device__ __forceinline__ v8f zero8() { v8f z = {}; return z; }

// build a 16-half WMMA fragment from two 16-byte chunks
__device__ __forceinline__ v16h ldfrag(const _Float16* p0, const _Float16* p1) {
    v8h lo = *(const v8h*)p0;
    v8h hi = *(const v8h*)p1;
    return __builtin_shufflevector(lo, hi, 0,1,2,3,4,5,6,7,8,9,10,11,12,13,14,15);
}

// cooperative stage of nvec*16B per thread from global to LDS
__device__ __forceinline__ void stage_b(const _Float16* __restrict__ gsrc,
                                        _Float16* __restrict__ ldst, int tid, int nvec) {
    const v8h* s = (const v8h*)gsrc;
    v8h*       d = (v8h*)ldst;
#if HAVE_ASYNC
    #pragma unroll
    for (int i = 0; i < 8; ++i) {
        if (i >= nvec) break;
        __builtin_amdgcn_global_load_async_to_lds_b128(
            (v4i*)(s + (size_t)tid * nvec + i),
            (v4i*)(d + (size_t)tid * nvec + i), 0, 0);
    }
#else
    #pragma unroll
    for (int i = 0; i < 8; ++i) {
        if (i >= nvec) break;
        d[(size_t)tid * nvec + i] = s[(size_t)tid * nvec + i];
    }
#endif
}

__device__ __forceinline__ void stage_wait() {
#if HAVE_ASYNC
#if __has_builtin(__builtin_amdgcn_s_wait_asynccnt)
    __builtin_amdgcn_s_wait_asynccnt(0);
#else
    asm volatile("s_wait_asynccnt 0x0" ::: "memory");
#endif
#endif
}

// ---------------- GCN kernels ----------------
__global__ void k_zero(float* __restrict__ p, int n) {
    int i = blockIdx.x * blockDim.x + threadIdx.x;
    if (i < n) p[i] = 0.0f;
}

__global__ void k_deg(const int* __restrict__ dst, float* __restrict__ deg) {
    int e = blockIdx.x * blockDim.x + threadIdx.x;
    if (e < NE) atomicAdd(&deg[dst[e]], 1.0f);
}

__global__ void k_dinv(float* __restrict__ deg) {
    int i = blockIdx.x * blockDim.x + threadIdx.x;
    if (i < NN) deg[i] = rsqrtf(deg[i] + 1.0f);  // +1 self loop; always > 0
}

// h0[n][f] = graph_x[n] @ conv1_W   (7 -> 16, bias applied after aggregation)
__global__ void k_lin1(const float* __restrict__ gx, const float* __restrict__ W,
                       float* __restrict__ h0) {
    int t = blockIdx.x * blockDim.x + threadIdx.x;
    if (t >= NN * 16) return;
    int n = t >> 4, f = t & 15;
    float s = 0.0f;
    #pragma unroll
    for (int c = 0; c < 7; ++c) s += gx[(size_t)n * 7 + c] * W[c * 16 + f];
    h0[t] = s;
}

// scatter with symmetric norm; idx < NE -> edge, else self loop node
__global__ void k_scatter(const int* __restrict__ src, const int* __restrict__ dst,
                          const float* __restrict__ dinv, const float* __restrict__ h,
                          float* __restrict__ agg) {
    long long t = (long long)blockIdx.x * blockDim.x + threadIdx.x;
    if (t >= (long long)(NE + NN) * 16) return;
    int f = (int)(t & 15);
    int idx = (int)(t >> 4);
    int s, d;
    if (idx < NE) { s = src[idx]; d = dst[idx]; }
    else          { s = d = idx - NE; }
    float nrm = dinv[s] * dinv[d];
    atomicAdd(&agg[(size_t)d * 16 + f], nrm * h[(size_t)s * 16 + f]);
}

// out[n] = gelu(agg1[n] + conv1_b) @ conv2_W   (16 -> 16)
__global__ void k_gelu_lin2(const float* __restrict__ agg1, const float* __restrict__ cb,
                            const float* __restrict__ W, float* __restrict__ out) {
    int n = blockIdx.x * blockDim.x + threadIdx.x;
    if (n >= NN) return;
    float g[16];
    #pragma unroll
    for (int c = 0; c < 16; ++c) g[c] = gelu_f(agg1[(size_t)n * 16 + c] + cb[c]);
    #pragma unroll
    for (int f = 0; f < 16; ++f) {
        float s = 0.0f;
        #pragma unroll
        for (int c = 0; c < 16; ++c) s += g[c] * W[c * 16 + f];
        out[(size_t)n * 16 + f] = s;
    }
}

// per-sample: gelu(agg2 + conv2_b) -> fc1 -> softmax over IAB
__global__ void k_fc1(const float* __restrict__ agg2, const float* __restrict__ cb,
                      const float* __restrict__ fcW, const float* __restrict__ fcb,
                      float* __restrict__ xi) {
    int b = blockIdx.x * blockDim.x + threadIdx.x;
    if (b >= B_N) return;
    float lg[10];
    #pragma unroll
    for (int j = 0; j < 10; ++j) lg[j] = fcb[j];
    for (int iab = 0; iab < IAB_N; ++iab) {
        #pragma unroll
        for (int f = 0; f < 16; ++f) {
            float hv = gelu_f(agg2[(size_t)(b * IAB_N + iab) * 16 + f] + cb[f]);
            const float* wrow = fcW + (iab * 16 + f) * 10;
            #pragma unroll
            for (int j = 0; j < 10; ++j) lg[j] += hv * wrow[j];
        }
    }
    float m = lg[0];
    #pragma unroll
    for (int j = 1; j < 10; ++j) m = fmaxf(m, lg[j]);
    float s = 0.0f;
    #pragma unroll
    for (int j = 0; j < 10; ++j) { lg[j] = expf(lg[j] - m); s += lg[j]; }
    float inv = 1.0f / s;
    #pragma unroll
    for (int j = 0; j < 10; ++j) xi[(size_t)b * 10 + j] = lg[j] * inv;
}

// ---------------- weight pack: f32 row-major -> f16 WMMA B-fragment order ----------------
// Layout (kt-major so one kt slice is contiguous for LDS staging):
// packed index ((kt*NT + nt)*32 + lane)*16 + e holds W[k][n] with
//   k = kt*32 + (lane>=16 ? 8 : 0) + (e&7) + (e>=8 ? 16 : 0),  n = nt*16 + (lane&15)   (zero padded)
__device__ __forceinline__ void packw(const float* __restrict__ W, _Float16* __restrict__ Wp,
                                      int idx, int ncols, int nvalid, int NT) {
    int e    = idx & 15;
    int lane = (idx >> 4) & 31;
    int rest = idx >> 9;
    int nt   = rest % NT;
    int kt   = rest / NT;
    int k = kt * 32 + (lane >> 4) * 8 + (e & 7) + (e >> 3) * 16;
    int n = nt * 16 + (lane & 15);
    float v = (k < 301 && n < nvalid) ? W[(size_t)k * ncols + n] : 0.0f;
    Wp[idx] = (_Float16)v;
}

__global__ void k_pack(const float* __restrict__ W1, const float* __restrict__ W2,
                       const float* __restrict__ W3,
                       _Float16* __restrict__ W1p, _Float16* __restrict__ W2p,
                       _Float16* __restrict__ W3p) {
    int t = blockIdx.x * blockDim.x + threadIdx.x;
    const int S1 = NT1 * KT * 32 * 16;   // 102400
    const int S3 = NT3 * KT * 32 * 16;   // 10240
    if (t < S1)               packw(W1, W1p, t,          301, 301, NT1);
    else if (t < 2 * S1)      packw(W2, W2p, t - S1,     301, 301, NT1);
    else if (t < 2 * S1 + S3) packw(W3, W3p, t - 2 * S1,  22,  22, NT3);
}

// ---------------- embedding MLP -> f16 A matrix [RR, 320] ----------------
__global__ void k_embed(const float* __restrict__ x, const float* __restrict__ eW1,
                        const float* __restrict__ eb1, const float* __restrict__ eW2,
                        const float* __restrict__ eb2, const float* __restrict__ xi,
                        _Float16* __restrict__ A0) {
    int t = blockIdx.x * blockDim.x + threadIdx.x;
    if (t >= RR * MAXUE_N) return;
    int r = t / MAXUE_N, u = t % MAXUE_N;
    int b = r / IAB_N, iab = r % IAB_N;
    const float* in = x + (size_t)b * XSTRIDE + 60 + iab * (MAXUE_N * TF_N) + u * TF_N;
    float f[6];
    #pragma unroll
    for (int c = 0; c < 6; ++c) f[c] = in[c];
    float h[30];
    #pragma unroll
    for (int j = 0; j < 30; ++j) {
        float s = eb1[j];
        #pragma unroll
        for (int c = 0; c < 6; ++c) s += f[c] * eW1[c * 30 + j];
        h[j] = fmaxf(s, 0.0f);
    }
    _Float16* dst = A0 + (size_t)r * KP + u * EMB_N;
    #pragma unroll
    for (int j = 0; j < 30; ++j) {
        float s = eb2[j];
        #pragma unroll
        for (int c = 0; c < 30; ++c) s += h[c] * eW2[c * 30 + j];
        dst[j] = (_Float16)s;
    }
    if (u == 0) {
        _Float16* row = A0 + (size_t)r * KP;
        row[300] = (_Float16)xi[r];
        #pragma unroll
        for (int c = 301; c < KP; ++c) row[c] = (_Float16)0.0f;
    }
}

// ---------------- fused head ----------------
// 8 waves x 16-row M-blocks = 128 rows/WG.  B weights staged per-kt into LDS
// (double buffered, async global->LDS when available); A for GEMM2/3 comes from
// the per-wave-private LDS activation tile (stride-padded against bank conflicts).
__global__ __launch_bounds__(HEAD_THREADS) void
k_head(const _Float16* __restrict__ A0,
       const _Float16* __restrict__ W1p, const _Float16* __restrict__ W2p,
       const _Float16* __restrict__ W3p,
       const float* __restrict__ b1, const float* __restrict__ b2,
       const float* __restrict__ b3,
       const float* __restrict__ ueidx, const float* __restrict__ xi,
       float* __restrict__ out) {
    extern __shared__ char smem[];
    _Float16* Hbuf = (_Float16*)smem;                       // ROWS_WG x HP halfs
    _Float16* Bst  = (_Float16*)(smem + HBYTES);            // 2 x BKHALFS halfs
    float*    Pbuf = (float*)(smem + HBYTES + 2 * BKBYTES); // ROWS_WG x 32 floats

    const int wg   = blockIdx.x;
    const int tid  = threadIdx.x;
    const int w    = tid >> 5;        // wave 0..7 -> M block
    const int lane = tid & 31;
    const int ln   = lane & 15;
    const int hi   = lane >> 4;

    const int rowBase = wg * ROWS_WG + w * 16;
    const int lrow    = w * 16;

    v8f acc[NT1];

    // ---- GEMM1: A0(global) x W1p(LDS-staged) -> gelu -> Hbuf ----
    stage_b(W1p, Bst, tid, 5);                        // kt=0 -> buf0
    #pragma unroll
    for (int i = 0; i < NT1; ++i) acc[i] = zero8();
    stage_wait(); __syncthreads();
    for (int kt = 0; kt < KT; ++kt) {
        const _Float16* cur = Bst + (kt & 1) * BKHALFS;
        if (kt + 1 < KT)
            stage_b(W1p + (size_t)(kt + 1) * BKHALFS, Bst + ((kt + 1) & 1) * BKHALFS, tid, 5);
        const _Float16* ab = A0 + (size_t)(rowBase + ln) * KP + kt * 32 + hi * 8;
        v16h a = ldfrag(ab, ab + 16);
        #pragma unroll
        for (int nt = 0; nt < NT1; ++nt) {
            const _Float16* bp = cur + ((size_t)(nt * 32 + lane) * 16);
            v16h bf = ldfrag(bp, bp + 8);
            acc[nt] = __builtin_amdgcn_wmma_f32_16x16x32_f16(
                false, a, false, bf, (short)0, acc[nt], false, false);
        }
        stage_wait(); __syncthreads();
    }
    #pragma unroll
    for (int nt = 0; nt < NT1; ++nt) {
        int n = nt * 16 + ln;
        #pragma unroll
        for (int r = 0; r < 8; ++r) {
            float v = acc[nt][r];
            v = (n < 301) ? gelu_f(v + b1[n]) : 0.0f;
            Hbuf[(size_t)(lrow + hi * 8 + r) * HP + n] = (_Float16)v;
        }
    }

    // ---- GEMM2: Hbuf(LDS) x W2p(LDS-staged) -> gelu -> Hbuf ----
    stage_b(W2p, Bst, tid, 5);
    #pragma unroll
    for (int i = 0; i < NT1; ++i) acc[i] = zero8();
    stage_wait(); __syncthreads();
    for (int kt = 0; kt < KT; ++kt) {
        const _Float16* cur = Bst + (kt & 1) * BKHALFS;
        if (kt + 1 < KT)
            stage_b(W2p + (size_t)(kt + 1) * BKHALFS, Bst + ((kt + 1) & 1) * BKHALFS, tid, 5);
        const _Float16* ab = &Hbuf[(size_t)(lrow + ln) * HP + kt * 32 + hi * 8];
        v16h a = ldfrag(ab, ab + 16);
        #pragma unroll
        for (int nt = 0; nt < NT1; ++nt) {
            const _Float16* bp = cur + ((size_t)(nt * 32 + lane) * 16);
            v16h bf = ldfrag(bp, bp + 8);
            acc[nt] = __builtin_amdgcn_wmma_f32_16x16x32_f16(
                false, a, false, bf, (short)0, acc[nt], false, false);
        }
        stage_wait(); __syncthreads();
    }
    #pragma unroll
    for (int nt = 0; nt < NT1; ++nt) {
        int n = nt * 16 + ln;
        #pragma unroll
        for (int r = 0; r < 8; ++r) {
            float v = acc[nt][r];
            v = (n < 301) ? gelu_f(v + b2[n]) : 0.0f;
            Hbuf[(size_t)(lrow + hi * 8 + r) * HP + n] = (_Float16)v;
        }
    }

    // ---- GEMM3: Hbuf x W3p (whole 20KB staged once; N=22 padded to 32) ----
    stage_b(W3p, Bst, tid, 5);                        // 256*5*8 = 10240 halfs = all of W3p
    v8f acc3[NT3];
    #pragma unroll
    for (int i = 0; i < NT3; ++i) acc3[i] = zero8();
    stage_wait(); __syncthreads();
    for (int kt = 0; kt < KT; ++kt) {
        const _Float16* ab = &Hbuf[(size_t)(lrow + ln) * HP + kt * 32 + hi * 8];
        v16h a = ldfrag(ab, ab + 16);
        #pragma unroll
        for (int nt = 0; nt < NT3; ++nt) {
            const _Float16* bp = Bst + ((size_t)((kt * NT3 + nt) * 32 + lane) * 16);
            v16h bf = ldfrag(bp, bp + 8);
            acc3[nt] = __builtin_amdgcn_wmma_f32_16x16x32_f16(
                false, a, false, bf, (short)0, acc3[nt], false, false);
        }
    }
    #pragma unroll
    for (int nt = 0; nt < NT3; ++nt) {
        int n = nt * 16 + ln;
        float bias = (n < 22) ? b3[n] : 0.0f;
        #pragma unroll
        for (int r = 0; r < 8; ++r)
            Pbuf[(size_t)(lrow + hi * 8 + r) * 32 + n] = acc3[nt][r] + bias;
    }
    __syncthreads();

    // ---- per-row epilogue: softmax(22) -> *ind -> /norm -> *x_iabs ----
    if (tid < ROWS_WG) {
        int gr = wg * ROWS_WG + tid;
        const float* pr = &Pbuf[(size_t)tid * 32];
        float m = pr[0];
        #pragma unroll
        for (int j = 1; j < 22; ++j) m = fmaxf(m, pr[j]);
        float e[22], s = 0.0f;
        #pragma unroll
        for (int j = 0; j < 22; ++j) { e[j] = expf(pr[j] - m); s += e[j]; }
        float inv = 1.0f / s;
        const float* ui = ueidx + (size_t)gr * (MAXUE_N * 2);
        float pv[22], nrm = 0.0f;
        #pragma unroll
        for (int j = 0; j < 22; ++j) {
            float ind = (j < 20) ? ui[j] : 1.0f;
            pv[j] = e[j] * inv * ind;
            nrm += pv[j];
        }
        float scale = xi[gr] / (nrm + 1e-8f);
        #pragma unroll
        for (int j = 0; j < 22; ++j) out[(size_t)gr * 22 + j] = pv[j] * scale;
    }
}

// ---------------- host launch ----------------
extern "C" void kernel_launch(void* const* d_in, const int* in_sizes, int n_in,
                              void* d_out, int out_size, void* d_ws, size_t ws_size,
                              hipStream_t stream) {
    (void)in_sizes; (void)n_in; (void)out_size; (void)ws_size;

    const float* x        = (const float*)d_in[0];
    const float* UEIdx    = (const float*)d_in[1];
    const float* graph_x  = (const float*)d_in[2];
    const int*   edge     = (const int*)  d_in[3];
    const float* eW1      = (const float*)d_in[4];
    const float* eb1      = (const float*)d_in[5];
    const float* eW2      = (const float*)d_in[6];
    const float* eb2      = (const float*)d_in[7];
    const float* c1W      = (const float*)d_in[8];
    const float* c1b      = (const float*)d_in[9];
    const float* c2W      = (const float*)d_in[10];
    const float* c2b      = (const float*)d_in[11];
    const float* fcW      = (const float*)d_in[12];
    const float* fcb      = (const float*)d_in[13];
    const float* sW1      = (const float*)d_in[14];
    const float* sb1      = (const float*)d_in[15];
    const float* sW2      = (const float*)d_in[16];
    const float* sb2      = (const float*)d_in[17];
    const float* sW3      = (const float*)d_in[18];
    const float* sb3      = (const float*)d_in[19];
    float* out = (float*)d_out;

    // workspace carve (256B aligned)
    uintptr_t p = (uintptr_t)d_ws;
    auto carve = [&p](size_t bytes) -> void* {
        void* r = (void*)p;
        p += (bytes + 255) & ~(size_t)255;
        return r;
    };
    float*     deg  = (float*)    carve((size_t)NN * 4);        // then dinv in-place
    float*     agg1 = (float*)    carve((size_t)NN * 16 * 4);
    float*     agg2 = (float*)    carve((size_t)NN * 16 * 4);
    float*     h0   = (float*)    carve((size_t)NN * 16 * 4);   // lin1 out, then lin2 out
    float*     xi   = (float*)    carve((size_t)RR * 4);        // x_iabs
    _Float16*  A0   = (_Float16*) carve((size_t)RR * KP * 2);   // head input, f16
    _Float16*  W1p  = (_Float16*) carve((size_t)KP * KP * 2);
    _Float16*  W2p  = (_Float16*) carve((size_t)KP * KP * 2);
    _Float16*  W3p  = (_Float16*) carve((size_t)KP * 32 * 2);

    const int T = 256;
    auto blocks = [](long long n, int t) { return (unsigned)((n + t - 1) / t); };

    const int* esrc = edge;
    const int* edst = edge + NE;

    // deg/agg1/agg2 are contiguous (all sizes multiples of 256B): one zero pass
    long long zcount = (long long)NN + 2LL * NN * 16;
    k_zero<<<blocks(zcount, T), T, 0, stream>>>(deg, (int)zcount);

    k_deg  <<<blocks(NE, T), T, 0, stream>>>(edst, deg);
    k_dinv <<<blocks(NN, T), T, 0, stream>>>(deg);
    k_lin1 <<<blocks((long long)NN * 16, T), T, 0, stream>>>(graph_x, c1W, h0);
    k_scatter<<<blocks((long long)(NE + NN) * 16, T), T, 0, stream>>>(esrc, edst, deg, h0, agg1);
    k_gelu_lin2<<<blocks(NN, T), T, 0, stream>>>(agg1, c1b, c2W, h0);
    k_scatter<<<blocks((long long)(NE + NN) * 16, T), T, 0, stream>>>(esrc, edst, deg, h0, agg2);
    k_fc1  <<<blocks(B_N, T), T, 0, stream>>>(agg2, c2b, fcW, fcb, xi);

    k_pack <<<blocks(2 * (NT1 * KT * 32 * 16) + (NT3 * KT * 32 * 16), T), T, 0, stream>>>(
        sW1, sW2, sW3, W1p, W2p, W3p);
    k_embed<<<blocks((long long)RR * MAXUE_N, T), T, 0, stream>>>(x, eW1, eb1, eW2, eb2, xi, A0);

    k_head <<<RR / ROWS_WG, HEAD_THREADS, SMEM_BYTES, stream>>>(
        A0, W1p, W2p, W3p, sb1, sb2, sb3, UEIdx, xi, out);
}
